// OneLayerGCNWithGlobalAdg_43808666419359
// MI455X (gfx1250) — compile-verified, matrix operand order checked by reading, not run on hardware
//
#include <hip/hip_runtime.h>
#include <hip/hip_bf16.h>
#include <math.h>

typedef __bf16 bf16_t;
typedef bf16_t v16bf __attribute__((ext_vector_type(16)));
typedef float  v8f   __attribute__((ext_vector_type(8)));

static constexpr int N_   = 32768;
static constexpr int IN_  = 1024;
static constexpr int OUT_ = 256;
static constexpr int E_   = 524288;
static constexpr int S_   = 64;
static constexpr int G_   = N_ / S_;

static constexpr int TILE_M = 64;
static constexpr int TILE_K = 32;           // one bf16 WMMA K-slab per tile
static constexpr int APAD   = TILE_K + 8;   // 40 bf16 = 20 DWORDs: conflict-min, 16B-aligned
static constexpr int BPAD   = TILE_K + 8;

// ---------------------------------------------------------------------------
// Zero-fill (h accumulator must start at 0 every call)
// ---------------------------------------------------------------------------
__global__ __launch_bounds__(256) void zero_f4(float4* __restrict__ p, int n4) {
    int i = blockIdx.x * 256 + threadIdx.x;
    if (i < n4) p[i] = make_float4(0.f, 0.f, 0.f, 0.f);
}

// ---------------------------------------------------------------------------
// Split f32 -> (hi, lo) bf16 pair: a ~= hi + lo, RNE both steps.
// ---------------------------------------------------------------------------
__device__ __forceinline__ void split_bf16(float a, bf16_t& hi, bf16_t& lo) {
    hi = (bf16_t)a;
    lo = (bf16_t)(a - (float)hi);
}

// Pack 8 bf16 (already in a local array) and store as one b128.
union Frag8 { bf16_t h[8]; uint4 u; };
union Frag16 { uint4 u[2]; v16bf v; };

// Load one 16-bit A/B fragment (16 bf16 = 8 VGPRs) from an LDS row.
// ISA layout: lanes 0-15 -> K 0-7 (chunk0), 16-23 (chunk1);
//             lanes 16-31 -> K 8-15, 24-31  => chunk offsets hi16*8 and 16+hi16*8.
__device__ __forceinline__ v16bf load_frag(const bf16_t* __restrict__ rowp, int hi16) {
    Frag16 f;
    f.u[0] = *(const uint4*)(rowp + hi16 * 8);
    f.u[1] = *(const uint4*)(rowp + 16 + hi16 * 8);
    return f.v;
}

// ---------------------------------------------------------------------------
// x = feat0 @ W via V_WMMA_F32_16X16X32_BF16 with bf16 hi/lo split products
// (hi*hi + hi*lo + lo*hi accumulated in f32 -> ~fp32 accuracy).
// Block: 256 threads = 8 waves, computes 64 rows x 256 cols.
// Wave w: row band (w>>1)*16, col half (w&1)*128 (8 col tiles of 16).
// feat0 = feat with anchor rows (global row % 64 == 0) zeroed.
// ---------------------------------------------------------------------------
__global__ __launch_bounds__(256) void gemm_wmma_bf16x3(const float* __restrict__ feat,
                                                        const float* __restrict__ Wm,
                                                        float* __restrict__ x) {
    __shared__ bf16_t Ah[TILE_M][APAD];
    __shared__ bf16_t Al[TILE_M][APAD];
    __shared__ bf16_t Bh[OUT_][BPAD];    // W tile transposed: [n][k]
    __shared__ bf16_t Bl[OUT_][BPAD];

    const int tid      = threadIdx.x;
    const int lane     = tid & 31;
    const int wave     = tid >> 5;
    const int rowBand  = (wave >> 1) * 16;      // 0,16,32,48
    const int colBase  = (wave & 1) * 128;      // 0 or 128
    const int halfLane = lane & 15;
    const int hi16     = lane >> 4;             // 0 / 1
    const int hi8      = hi16 * 8;              // C/D: lanes 16-31 hold M=j+8
    const int row0     = blockIdx.x * TILE_M;

    v8f acc[8];
#pragma unroll
    for (int t = 0; t < 8; ++t) acc[t] = (v8f){0.f,0.f,0.f,0.f,0.f,0.f,0.f,0.f};

    for (int k0 = 0; k0 < IN_; k0 += TILE_K) {
        // ---- stage A tile: 64 rows x 32 k, 8 consecutive f32 per thread.
        {
            int v  = tid * 8;                   // 0..2047
            int r  = v >> 5;                    // row 0..63
            int c  = v & 31;                    // 0,8,16,24
            Frag8 fh, fl;
            if (r == 0) {                       // anchor row of this subgraph -> 0
#pragma unroll
                for (int i = 0; i < 8; ++i) { fh.h[i] = (bf16_t)0.f; fl.h[i] = (bf16_t)0.f; }
            } else {
                const float* ap = feat + (size_t)(row0 + r) * IN_ + k0 + c;
                float4 a0 = *(const float4*)(ap);
                float4 a1 = *(const float4*)(ap + 4);
                float av[8] = {a0.x,a0.y,a0.z,a0.w,a1.x,a1.y,a1.z,a1.w};
#pragma unroll
                for (int i = 0; i < 8; ++i) split_bf16(av[i], fh.h[i], fl.h[i]);
            }
            *(uint4*)&Ah[r][c] = fh.u;          // byte offset c*2 in {0,16,32,48}: aligned
            *(uint4*)&Al[r][c] = fl.u;
        }
        // ---- stage B tile transposed: thread owns column n=tid, k in chunks of 8.
        {
#pragma unroll
            for (int kk = 0; kk < TILE_K; kk += 8) {
                Frag8 fh, fl;
#pragma unroll
                for (int i = 0; i < 8; ++i) {
                    float w = Wm[(size_t)(k0 + kk + i) * OUT_ + tid];  // coalesced across tid
                    split_bf16(w, fh.h[i], fl.h[i]);
                }
                *(uint4*)&Bh[tid][kk] = fh.u;
                *(uint4*)&Bl[tid][kk] = fl.u;
            }
        }
        __syncthreads();

        // ---- one K=32 slab: 3 WMMAs per 16x16 col tile (hi*hi + hi*lo + lo*hi)
        const int arow = rowBand + halfLane;
        v16bf a_h = load_frag(&Ah[arow][0], hi16);
        v16bf a_l = load_frag(&Al[arow][0], hi16);
#pragma unroll
        for (int t = 0; t < 8; ++t) {
            const int brow = colBase + t * 16 + halfLane;
            v16bf b_h = load_frag(&Bh[brow][0], hi16);
            v16bf b_l = load_frag(&Bl[brow][0], hi16);
            acc[t] = __builtin_amdgcn_wmma_f32_16x16x32_bf16(
                false, a_h, false, b_h, (short)0, acc[t], false, false);
            acc[t] = __builtin_amdgcn_wmma_f32_16x16x32_bf16(
                false, a_h, false, b_l, (short)0, acc[t], false, false);
            acc[t] = __builtin_amdgcn_wmma_f32_16x16x32_bf16(
                false, a_l, false, b_h, (short)0, acc[t], false, false);
        }
        __syncthreads();
    }

    // ---- store D: VGPR j holds M = rowBand + hi8 + j, N = n0 + halfLane
#pragma unroll
    for (int t = 0; t < 8; ++t) {
        int n0 = colBase + t * 16;
#pragma unroll
        for (int j = 0; j < 8; ++j) {
            int gr = row0 + rowBand + hi8 + j;
            x[(size_t)gr * OUT_ + n0 + halfLane] = acc[t][j];
        }
    }
}

// ---------------------------------------------------------------------------
// SpMM: h[dst] += w_e * x[src].  One thread per (edge, 4-float chunk).
// ---------------------------------------------------------------------------
__global__ __launch_bounds__(256) void spmm_atomic(const float* __restrict__ x,
                                                   const int* __restrict__ src,
                                                   const int* __restrict__ dst,
                                                   const float* __restrict__ ew,
                                                   float* __restrict__ h) {
    int tid = blockIdx.x * 256 + threadIdx.x;
    int e   = tid >> 6;                 // 64 threads per edge
    if (e >= E_) return;
    int c4  = (tid & 63) * 4;
    int s   = src[e];
    int d   = dst[e];
    float w = ew[e];
    float4 xv = *(const float4*)(x + (size_t)s * OUT_ + c4);
    float* hp = h + (size_t)d * OUT_ + c4;
    unsafeAtomicAdd(hp + 0, w * xv.x);
    unsafeAtomicAdd(hp + 1, w * xv.y);
    unsafeAtomicAdd(hp + 2, w * xv.z);
    unsafeAtomicAdd(hp + 3, w * xv.w);
}

// ---------------------------------------------------------------------------
// hb = relu(h + bias); out = l2norm(hb) per row; h <- hb (in place, for pool).
// One wave32 per 256-wide row, 8 rows per block.
// ---------------------------------------------------------------------------
__global__ __launch_bounds__(256) void relu_norm_rows(float* __restrict__ h,
                                                      const float* __restrict__ bias,
                                                      float* __restrict__ out) {
    int row  = blockIdx.x * 8 + (threadIdx.x >> 5);
    int lane = threadIdx.x & 31;
    float v[8];
    float ss = 0.f;
#pragma unroll
    for (int j = 0; j < 8; ++j) {
        int c = lane + j * 32;
        float t = h[(size_t)row * OUT_ + c] + bias[c];
        t = fmaxf(t, 0.f);
        v[j] = t;
        ss  += t * t;
    }
#pragma unroll
    for (int off = 16; off > 0; off >>= 1)
        ss += __shfl_xor(ss, off, 32);
    float inv = 1.f / fmaxf(sqrtf(ss), 1e-12f);
#pragma unroll
    for (int j = 0; j < 8; ++j) {
        int c = lane + j * 32;
        out[(size_t)row * OUT_ + c] = v[j] * inv;
        h[(size_t)row * OUT_ + c]   = v[j];
    }
}

// ---------------------------------------------------------------------------
// pool[g] = l2norm(mean over S rows of hb).  Block per subgraph, 256 threads.
// ---------------------------------------------------------------------------
__global__ __launch_bounds__(256) void pool_norm(const float* __restrict__ h,
                                                 float* __restrict__ out2) {
    __shared__ float red[256];
    int g = blockIdx.x;
    int c = threadIdx.x;
    float s = 0.f;
    for (int i = 0; i < S_; ++i)
        s += h[(size_t)(g * S_ + i) * OUT_ + c];
    float p = s * (1.f / (float)S_);
    red[c] = p * p;
    __syncthreads();
    for (int off = 128; off > 0; off >>= 1) {
        if (c < off) red[c] += red[c + off];
        __syncthreads();
    }
    float inv = 1.f / fmaxf(sqrtf(red[0]), 1e-12f);
    out2[(size_t)g * OUT_ + c] = p * inv;
}

// ---------------------------------------------------------------------------
// anchor_out[g] = l2norm(relu(feat[g*S] @ W + bias)).  Block per graph.
// ---------------------------------------------------------------------------
__global__ __launch_bounds__(256) void anchor_norm(const float* __restrict__ feat,
                                                   const float* __restrict__ Wm,
                                                   const float* __restrict__ bias,
                                                   float* __restrict__ out3) {
    __shared__ float af[IN_];
    __shared__ float red[256];
    int g = blockIdx.x;
    int c = threadIdx.x;
    const float* frow = feat + (size_t)g * S_ * IN_;
    for (int k = c; k < IN_; k += 256) af[k] = frow[k];
    __syncthreads();
    float acc = 0.f;
    for (int k = 0; k < IN_; ++k)
        acc = fmaf(af[k], Wm[(size_t)k * OUT_ + c], acc);
    float v = fmaxf(acc + bias[c], 0.f);
    red[c] = v * v;
    __syncthreads();
    for (int off = 128; off > 0; off >>= 1) {
        if (c < off) red[c] += red[c + off];
        __syncthreads();
    }
    float inv = 1.f / fmaxf(sqrtf(red[0]), 1e-12f);
    out3[(size_t)g * OUT_ + c] = v * inv;
}

// ---------------------------------------------------------------------------
extern "C" void kernel_launch(void* const* d_in, const int* in_sizes, int n_in,
                              void* d_out, int out_size, void* d_ws, size_t ws_size,
                              hipStream_t stream) {
    (void)in_sizes; (void)n_in; (void)out_size; (void)ws_size;

    const float* feat = (const float*)d_in[0];
    const int*   src  = (const int*)  d_in[1];
    const int*   dst  = (const int*)  d_in[2];
    const float* ew   = (const float*)d_in[3];
    const float* Wm   = (const float*)d_in[4];
    const float* bias = (const float*)d_in[5];

    float* out        = (float*)d_out;
    float* out_h      = out;                                // [N, OUT]
    float* out_pool   = out + (size_t)N_ * OUT_;            // [G, OUT]
    float* out_anchor = out_pool + (size_t)G_ * OUT_;       // [G, OUT]

    float* x = (float*)d_ws;                                // [N, OUT]
    float* h = x + (size_t)N_ * OUT_;                       // [N, OUT]

    // 1) h = 0
    {
        int n4 = (N_ * OUT_) / 4;
        zero_f4<<<(n4 + 255) / 256, 256, 0, stream>>>((float4*)h, n4);
    }
    // 2) x = feat0 @ W  (bf16 split-product WMMA, fp32-grade accuracy)
    gemm_wmma_bf16x3<<<N_ / TILE_M, 256, 0, stream>>>(feat, Wm, x);
    // 3) h[dst] += w_e * x[src]
    spmm_atomic<<<(E_ * 64) / 256, 256, 0, stream>>>(x, src, dst, ew, h);
    // 4) hb = relu(h+bias); out_h = l2norm(hb); h <- hb
    relu_norm_rows<<<N_ / 8, 256, 0, stream>>>(h, bias, out_h);
    // 5) out_pool = l2norm(mean_S(hb))
    pool_norm<<<G_, 256, 0, stream>>>(h, out_pool);
    // 6) out_anchor = l2norm(relu(feat[anchor] @ W + bias))
    anchor_norm<<<G_, 256, 0, stream>>>(feat, Wm, bias, out_anchor);
}
